// NewGRU_78795470012672
// MI455X (gfx1250) — compile-verified, matrix operand order checked by reading, not run on hardware
//
#include <hip/hip_runtime.h>
#include <stdint.h>
#include <stddef.h>

// Problem sizes (fixed by the reference)
#define B_   64
#define T_   2048
#define D_   256
#define U_   256
#define NWG  16          // persistent workgroups in the sequential kernel

typedef __attribute__((ext_vector_type(16))) __bf16 bf16x16;
typedef __attribute__((ext_vector_type(8)))  float  f32x8;

union Frag {
  bf16x16        v;
  uint4          q[2];
  unsigned short s[16];
};
union Half8 {                 // 8 bf16 = 16 bytes
  unsigned short s[8];
  uint4          q;
};

// native bf16 converts (v_cvt_pk_bf16_f32 pairs on gfx1250)
__device__ __forceinline__ unsigned short f2bf(float f) {
  return __builtin_bit_cast(unsigned short, (__bf16)f);
}
__device__ __forceinline__ float bf2f(unsigned short s) {
  return (float)__builtin_bit_cast(__bf16, s);
}
__device__ __forceinline__ float sigmoidf_(float x) { return 1.0f / (1.0f + __expf(-x)); }

// ---- fragment loaders -------------------------------------------------------
// 16-bit A/B fragment (16x32), ISA layout: lane L (0..15) holds row L with
// K in {k0..k0+7} (q[0]) and {k0+16..k0+23} (q[1]); lanes 16..31 use k0+8.
__device__ __forceinline__ Frag load_frag_bf16(const unsigned short* base, int row, int k0) {
  Frag a;
  const uint4* p = (const uint4*)(base + row * 256 + k0);
  a.q[0] = p[0];
  a.q[1] = p[2];   // +16 elements = +32 bytes
  return a;
}
__device__ __forceinline__ Frag load_frag_lds(const unsigned short* base, int nl, int k0) {
  Frag b;
  const uint4* p = (const uint4*)(base + nl * 256 + k0);
  b.q[0] = p[0];
  b.q[1] = p[2];
  return b;
}
__device__ __forceinline__ Frag load_frag_f32(const float* rowptr, int k0) {
  Frag a;
  const float4* p0 = (const float4*)(rowptr + k0);
  float4 f0 = p0[0], f1 = p0[1];
  const float4* p1 = (const float4*)(rowptr + k0 + 16);
  float4 f2 = p1[0], f3 = p1[1];
  a.v[0]  = (__bf16)f0.x;  a.v[1]  = (__bf16)f0.y;  a.v[2]  = (__bf16)f0.z;  a.v[3]  = (__bf16)f0.w;
  a.v[4]  = (__bf16)f1.x;  a.v[5]  = (__bf16)f1.y;  a.v[6]  = (__bf16)f1.z;  a.v[7]  = (__bf16)f1.w;
  a.v[8]  = (__bf16)f2.x;  a.v[9]  = (__bf16)f2.y;  a.v[10] = (__bf16)f2.z;  a.v[11] = (__bf16)f2.w;
  a.v[12] = (__bf16)f3.x;  a.v[13] = (__bf16)f3.y;  a.v[14] = (__bf16)f3.z;  a.v[15] = (__bf16)f3.w;
  return a;
}
__device__ __forceinline__ f32x8 wmma_bf16(const Frag& a, const Frag& b, f32x8 c) {
  return __builtin_amdgcn_wmma_f32_16x16x32_bf16(false, a.v, false, b.v, (short)0, c, false, false);
}

// ---- device-wide sense barrier for NWG resident workgroups ------------------
__device__ __forceinline__ void grid_barrier(int* cnt, int* gen, int nwg) {
  __threadfence();
  __syncthreads();
  if (threadIdx.x == 0) {
    int g = __hip_atomic_load(gen, __ATOMIC_RELAXED, __HIP_MEMORY_SCOPE_AGENT);
    int arrived = __hip_atomic_fetch_add(cnt, 1, __ATOMIC_ACQ_REL, __HIP_MEMORY_SCOPE_AGENT);
    if (arrived == nwg - 1) {
      __hip_atomic_store(cnt, 0, __ATOMIC_RELAXED, __HIP_MEMORY_SCOPE_AGENT);
      __hip_atomic_store(gen, g + 1, __ATOMIC_RELEASE, __HIP_MEMORY_SCOPE_AGENT);
    } else {
      while (__hip_atomic_load(gen, __ATOMIC_ACQUIRE, __HIP_MEMORY_SCOPE_AGENT) == g) {
        __builtin_amdgcn_s_sleep(1);
      }
    }
  }
  __syncthreads();
}

// ---- weight prep: fp32 [K][N] -> bf16 transposed [N][K] ---------------------
// Wt matrix order: 0=Whu 1=Whr 2=Wro 3=Wxu 4=Wxr 5=Wxo, each 256x256 bf16.
__global__ void prep_weights_kernel(const float* __restrict__ Whu, const float* __restrict__ Whr,
                                    const float* __restrict__ Wro, const float* __restrict__ Wxu,
                                    const float* __restrict__ Wxr, const float* __restrict__ Wxo,
                                    unsigned short* __restrict__ Wt) {
  int idx = blockIdx.x * blockDim.x + threadIdx.x;   // 0 .. 6*65536-1
  int m = idx >> 16;
  int r = idx & 65535;
  int nrow = r >> 8, k = r & 255;
  const float* src = (m == 0) ? Whu : (m == 1) ? Whr : (m == 2) ? Wro
                    : (m == 3) ? Wxu : (m == 4) ? Wxr : Wxo;
  Wt[idx] = f2bf(src[k * 256 + nrow]);
}

// Xproj layout: tile-major, fragment-order.
// Xg[ (t*64 + (mt*16+nt)) * 256 + lane*8 + g ]  (16B contiguous per lane)
__device__ __forceinline__ size_t xidx(int t, int mt, int nt, int lane) {
  return ((size_t)t * 64 + (size_t)(mt * 16 + nt)) * 256 + (size_t)lane * 8;
}

// ---- parallel x-projection: (x_t @ W_xg + b_g) in bf16, fragment layout -----
__global__ void gru_xproj_kernel(const float* __restrict__ x,
                                 const unsigned short* __restrict__ Wt,
                                 const float* __restrict__ b_u, const float* __restrict__ b_r,
                                 const float* __restrict__ b_o,
                                 unsigned short* __restrict__ Xu, unsigned short* __restrict__ Xr,
                                 unsigned short* __restrict__ Xo) {
  const int tid = threadIdx.x, lane = tid & 31, wave = tid >> 5; // 8 waves/block
  const int bid = blockIdx.x;
  const int t = bid / 24, sub = bid % 24;
  const int tile = sub * 8 + wave;        // 0..191 tiles per t
  const int mt = tile & 3;                // 4 row tiles (B=64)
  const int nn = tile >> 2;               // 0..47 = gate*16 + nt
  const int gate = nn >> 4;
  const int nt = nn & 15;
  const int hi = lane >> 4, nl = lane & 15;
  const int n = nt * 16 + nl;
  const unsigned short* Wg = Wt + (size_t)(3 + gate) * 65536;
  const float* bias = (gate == 0) ? b_u : (gate == 1) ? b_r : b_o;
  unsigned short* Xg = (gate == 0) ? Xu : (gate == 1) ? Xr : Xo;

  const int arow = mt * 16 + nl;                          // batch row per lane
  const float* xp = x + ((size_t)arow * T_ + t) * D_;     // x is [B][T][D]
  f32x8 acc0, acc1;
  float bv = bias[n];
#pragma unroll
  for (int g = 0; g < 8; ++g) { acc0[g] = bv; acc1[g] = 0.0f; }
#pragma unroll
  for (int f = 0; f < 4; ++f) {           // two independent 4-deep WMMA chains
    int k0 = f * 32 + 8 * hi;
    int k1 = (f + 4) * 32 + 8 * hi;
    Frag a0 = load_frag_f32(xp, k0);
    Frag b0 = load_frag_bf16(Wg, n, k0);
    acc0 = wmma_bf16(a0, b0, acc0);
    Frag a1 = load_frag_f32(xp, k1);
    Frag b1 = load_frag_bf16(Wg, n, k1);
    acc1 = wmma_bf16(a1, b1, acc1);
  }
  Half8 hout;
#pragma unroll
  for (int g = 0; g < 8; ++g) hout.s[g] = f2bf(acc0[g] + acc1[g]);
  *(uint4*)(Xg + xidx(t, mt, nt, lane)) = hout.q;        // one b128 store
}

// ---- persistent sequential recurrence ---------------------------------------
template <bool FAST>
__global__ void gru_seq_kernel(const float* __restrict__ x,
                               const unsigned short* __restrict__ Xu,
                               const unsigned short* __restrict__ Xr,
                               const unsigned short* __restrict__ Xo,
                               const unsigned short* __restrict__ Wt,
                               const float* __restrict__ b_u, const float* __restrict__ b_r,
                               const float* __restrict__ b_o,
                               unsigned short* __restrict__ hbf,
                               unsigned short* __restrict__ rbf,
                               int* __restrict__ bar, float* __restrict__ out) {
  constexpr int NMAT = FAST ? 3 : 6;                 // Whu,Whr,Wro (+Wxu,Wxr,Wxo)
  __shared__ alignas(16) unsigned short ldsW[NMAT * 16 * 256];   // 24KB / 48KB

  const int tid = threadIdx.x;
  const int lane = tid & 31;
  const int wave = tid >> 5;      // 4 waves -> mt
  const int blk = blockIdx.x;     // 16 WGs -> nt
  const int mt = wave;
  const int hi = lane >> 4;
  const int nl = lane & 15;
  const int n = blk * 16 + nl;
  const int arow = mt * 16 + nl;

  // stage this WG's 16-column weight slices into LDS with the CDNA5 async
  // global->LDS path (ASYNCcnt), 16B per op, contiguous 8KB per matrix.
  for (int m = 0; m < NMAT; ++m) {
    const unsigned short* gsrc = Wt + (size_t)m * 65536 + (size_t)blk * 16 * 256;
#pragma unroll
    for (int c = 0; c < 4; ++c) {                  // 512 chunks / 128 threads
      int i = tid + c * 128;
      unsigned int ldsoff = (unsigned int)(m * 8192 + i * 16);   // byte offset
      const unsigned short* gp = gsrc + (size_t)i * 8;
      asm volatile("global_load_async_to_lds_b128 %0, %1, off"
                   :: "v"(ldsoff), "v"(gp)
                   : "memory");
    }
  }
  asm volatile("s_wait_asynccnt 0" ::: "memory");

  // h tile lives in registers (fp32); only the bf16 mirror is shared.
  // C/D layout: VGPR g, lane -> (m = g + 8*hi, n)
  int midx[8];
  float hreg[8];
#pragma unroll
  for (int g = 0; g < 8; ++g) {
    int M = mt * 16 + g + 8 * hi;
    midx[g] = M * 256 + n;
    hreg[g] = 0.0f;
    hbf[midx[g]] = 0;
  }
  float bu_s = 0.0f, br_s = 0.0f, bo_s = 0.0f;
  if (!FAST) { bu_s = b_u[n]; br_s = b_r[n]; bo_s = b_o[n]; }
  grid_barrier(bar, bar + 1, NWG);

  for (int t = 0; t < T_; ++t) {
    // ---- stage 1: u, r gates ----
    Frag hA[8];
#pragma unroll
    for (int f = 0; f < 8; ++f) hA[f] = load_frag_bf16(hbf, arow, f * 32 + 8 * hi);

    Frag xA[8];
    if (!FAST) {
      const float* xp = x + ((size_t)arow * T_ + t) * D_;
#pragma unroll
      for (int f = 0; f < 8; ++f) xA[f] = load_frag_f32(xp, f * 32 + 8 * hi);
      if (t + 1 < T_) {  // prefetch next step's x row
        const float* xpn = x + ((size_t)arow * T_ + (t + 1)) * D_;
        __builtin_prefetch(xpn, 0, 3);
        __builtin_prefetch(xpn + 128, 0, 3);
      }
    }

    f32x8 accU0, accU1, accR0, accR1;
    if (FAST) {
      Half8 pu, pr;
      pu.q = *(const uint4*)(Xu + xidx(t, mt, blk, lane));   // one b128 load each
      pr.q = *(const uint4*)(Xr + xidx(t, mt, blk, lane));
#pragma unroll
      for (int g = 0; g < 8; ++g) {
        accU0[g] = bf2f(pu.s[g]); accR0[g] = bf2f(pr.s[g]);
        accU1[g] = 0.0f;          accR1[g] = 0.0f;
      }
      if (t + 1 < T_) {  // prefetch next step's X tiles across the barriers
        __builtin_prefetch(Xu + xidx(t + 1, mt, blk, lane), 0, 3);
        __builtin_prefetch(Xr + xidx(t + 1, mt, blk, lane), 0, 3);
        __builtin_prefetch(Xo + xidx(t + 1, mt, blk, lane), 0, 3);
      }
    } else {
#pragma unroll
      for (int g = 0; g < 8; ++g) {
        accU0[g] = bu_s; accR0[g] = br_s;
        accU1[g] = 0.0f; accR1[g] = 0.0f;
      }
    }
    // two independent 4-deep chains per gate (K halves)
#pragma unroll
    for (int f = 0; f < 4; ++f) {
      int k0 = f * 32 + 8 * hi;
      int k1 = (f + 4) * 32 + 8 * hi;
      Frag bu0 = load_frag_lds(ldsW + 0, nl, k0);        // Whu
      accU0 = wmma_bf16(hA[f], bu0, accU0);
      Frag bu1 = load_frag_lds(ldsW + 0, nl, k1);
      accU1 = wmma_bf16(hA[f + 4], bu1, accU1);
      Frag br0 = load_frag_lds(ldsW + 4096, nl, k0);     // Whr
      accR0 = wmma_bf16(hA[f], br0, accR0);
      Frag br1 = load_frag_lds(ldsW + 4096, nl, k1);
      accR1 = wmma_bf16(hA[f + 4], br1, accR1);
    }
    if (!FAST) {
#pragma unroll
      for (int f = 0; f < 4; ++f) {
        int k0 = f * 32 + 8 * hi;
        int k1 = (f + 4) * 32 + 8 * hi;
        Frag bu0 = load_frag_lds(ldsW + 12288, nl, k0);  // Wxu
        accU0 = wmma_bf16(xA[f], bu0, accU0);
        Frag bu1 = load_frag_lds(ldsW + 12288, nl, k1);
        accU1 = wmma_bf16(xA[f + 4], bu1, accU1);
        Frag br0 = load_frag_lds(ldsW + 16384, nl, k0);  // Wxr
        accR0 = wmma_bf16(xA[f], br0, accR0);
        Frag br1 = load_frag_lds(ldsW + 16384, nl, k1);
        accR1 = wmma_bf16(xA[f + 4], br1, accR1);
      }
    }
    float ug[8];
#pragma unroll
    for (int g = 0; g < 8; ++g) {
      ug[g] = sigmoidf_(accU0[g] + accU1[g]);
      float rg = sigmoidf_(accR0[g] + accR1[g]);
      rbf[midx[g]] = f2bf(rg);
    }
    grid_barrier(bar, bar + 1, NWG);   // r visible device-wide

    // ---- stage 2: o = tanh(xWxo + r Wro + bo), h update ----
    Frag rA[8];
#pragma unroll
    for (int f = 0; f < 8; ++f) rA[f] = load_frag_bf16(rbf, arow, f * 32 + 8 * hi);

    f32x8 accO0, accO1;
    if (FAST) {
      Half8 po;
      po.q = *(const uint4*)(Xo + xidx(t, mt, blk, lane));
#pragma unroll
      for (int g = 0; g < 8; ++g) { accO0[g] = bf2f(po.s[g]); accO1[g] = 0.0f; }
    } else {
#pragma unroll
      for (int g = 0; g < 8; ++g) { accO0[g] = bo_s; accO1[g] = 0.0f; }
    }
#pragma unroll
    for (int f = 0; f < 4; ++f) {
      int k0 = f * 32 + 8 * hi;
      int k1 = (f + 4) * 32 + 8 * hi;
      Frag bo0 = load_frag_lds(ldsW + 8192, nl, k0);     // Wro
      accO0 = wmma_bf16(rA[f], bo0, accO0);
      Frag bo1 = load_frag_lds(ldsW + 8192, nl, k1);
      accO1 = wmma_bf16(rA[f + 4], bo1, accO1);
    }
    if (!FAST) {
#pragma unroll
      for (int f = 0; f < 4; ++f) {
        int k0 = f * 32 + 8 * hi;
        int k1 = (f + 4) * 32 + 8 * hi;
        Frag bo0 = load_frag_lds(ldsW + 20480, nl, k0);  // Wxo
        accO0 = wmma_bf16(xA[f], bo0, accO0);
        Frag bo1 = load_frag_lds(ldsW + 20480, nl, k1);
        accO1 = wmma_bf16(xA[f + 4], bo1, accO1);
      }
    }
#pragma unroll
    for (int g = 0; g < 8; ++g) {
      float o = tanhf(accO0[g] + accO1[g]);
      float hn = (1.0f - ug[g]) * hreg[g] + ug[g] * o;
      hreg[g] = hn;
      hbf[midx[g]] = f2bf(hn);
      if (t == T_ - 1) out[midx[g]] = hn;
    }
    grid_barrier(bar, bar + 1, NWG);   // h visible device-wide
  }
}

// ---- host side --------------------------------------------------------------
extern "C" void kernel_launch(void* const* d_in, const int* in_sizes, int n_in,
                              void* d_out, int out_size, void* d_ws, size_t ws_size,
                              hipStream_t stream) {
  const float* x    = (const float*)d_in[0];
  const float* W_xu = (const float*)d_in[1];
  const float* W_hu = (const float*)d_in[2];
  const float* b_u  = (const float*)d_in[3];
  const float* W_xr = (const float*)d_in[4];
  const float* W_hr = (const float*)d_in[5];
  const float* b_r  = (const float*)d_in[6];
  const float* W_xo = (const float*)d_in[7];
  const float* W_ro = (const float*)d_in[8];
  const float* b_o  = (const float*)d_in[9];

  char* ws = (char*)d_ws;
  // layout: [0,256) barrier | hbf 32KB | rbf 32KB | Wt 768KB | Xproj @1MB
  int*            bar = (int*)ws;
  unsigned short* hbf = (unsigned short*)(ws + 256);
  unsigned short* rbf = (unsigned short*)(ws + 256 + 32768);
  unsigned short* Wt  = (unsigned short*)(ws + 256 + 32768 + 32768);
  const size_t xbase = 1u << 20;
  unsigned short* Xu = (unsigned short*)(ws + xbase);
  unsigned short* Xr = Xu + (size_t)T_ * 64 * 256;
  unsigned short* Xo = Xr + (size_t)T_ * 64 * 256;
  const size_t needFast = xbase + 3ull * T_ * 64 * 256 * 2;   // ~203 MB
  const bool fast = (ws_size >= needFast);

  hipMemsetAsync(d_ws, 0, 256, stream);   // barrier state
  prep_weights_kernel<<<(6 * 65536) / 256, 256, 0, stream>>>(W_hu, W_hr, W_ro,
                                                             W_xu, W_xr, W_xo, Wt);
  if (fast) {
    gru_xproj_kernel<<<T_ * 24, 256, 0, stream>>>(x, Wt, b_u, b_r, b_o, Xu, Xr, Xo);
    gru_seq_kernel<true><<<NWG, 128, 0, stream>>>(x, Xu, Xr, Xo, Wt, b_u, b_r, b_o,
                                                  hbf, rbf, bar, (float*)d_out);
  } else {
    gru_seq_kernel<false><<<NWG, 128, 0, stream>>>(x, nullptr, nullptr, nullptr, Wt,
                                                   b_u, b_r, b_o,
                                                   hbf, rbf, bar, (float*)d_out);
  }
}
